// ImageSelfAttention_58609123721899
// MI455X (gfx1250) — matrix operand order repaired and checked by reading, not compiled, not run on hardware
//
#include <hip/hip_runtime.h>
#include <hip/hip_bf16.h>
#include <cstdint>

// ---- problem constants ----
#define BB    4
#define CIN   256
#define HH_   48
#define WW_   48
#define SLEN  2304          // 48*48
#define PTOT  9216          // B*SLEN
#define HEADS 8
#define DH    64
#define HID   512           // HEADS*DH
#define QKVO  1536          // 3*HID
#define NT_QKV 96           // QKVO/16
#define KT_QKV 8            // CIN/32
#define NT_OUT 16           // CIN/16
#define KT_OUT 16           // HID/32

typedef __attribute__((ext_vector_type(16))) __bf16 v16bf;
typedef __attribute__((ext_vector_type(8)))  __bf16 v8bf;
typedef __attribute__((ext_vector_type(8)))  float  v8f;

__device__ __forceinline__ v8f wmma_bf16(v16bf a, v16bf b, v8f c) {
  return __builtin_amdgcn_wmma_f32_16x16x32_bf16(
      false, a, false, b, (short)0, c, false, false);
}

// ---- CDNA5 async global->LDS path (guarded; falls back to plain copy) ----
#if defined(__has_builtin)
# if __has_builtin(__builtin_amdgcn_global_load_async_to_lds_b128)
#  define HAVE_ASYNC_LDS 1
# endif
# if __has_builtin(__builtin_amdgcn_s_wait_asynccnt)
#  define HAVE_WAIT_ASYNC_BUILTIN 1
# endif
#endif

typedef int v4i __attribute__((vector_size(16)));
typedef __attribute__((address_space(1))) v4i as1_v4i;
typedef __attribute__((address_space(3))) v4i as3_v4i;

__device__ __forceinline__ as1_v4i* as_global_v4(const void* p) {
  return (as1_v4i*)(unsigned long long)(size_t)p;
}
__device__ __forceinline__ as3_v4i* as_lds_v4(const void* p) {
  return (as3_v4i*)(unsigned)(size_t)p;
}
__device__ __forceinline__ void wait_async0() {
#if defined(HAVE_ASYNC_LDS)
# if defined(HAVE_WAIT_ASYNC_BUILTIN)
  __builtin_amdgcn_s_wait_asynccnt(0);
# else
  asm volatile("s_wait_asynccnt 0" ::: "memory");
# endif
#endif
}

// fragment-layout helpers: channel-sub (0..31) -> (half, elem) of A/B frag
// half = bit3 of sub; elem = (sub&7) | ((sub>>4)<<3)

// ---------------------------------------------------------------------------
// 1) Channel LayerNorm per position; emits xn in packed A-fragment layout:
//    xn[((tm*KT + kt)*32 + lane)*16 + e], tm = p/16
// ---------------------------------------------------------------------------
__global__ void __launch_bounds__(256) ln_kernel(const float* __restrict__ x,
                                                 const float* __restrict__ g,
                                                 __bf16* __restrict__ xnp) {
  int lane = threadIdx.x & 31;
  int wv   = threadIdx.x >> 5;
  int p    = blockIdx.x * 8 + wv;           // 0..PTOT-1
  int b = p / SLEN, sp = p % SLEN;
  const float* xp = x + (size_t)b * CIN * SLEN + sp;
  float vals[8], s = 0.f, s2 = 0.f;
#pragma unroll
  for (int i = 0; i < 8; ++i) {
    float v = xp[(size_t)(lane + i * 32) * SLEN];
    vals[i] = v; s += v; s2 += v * v;
  }
#pragma unroll
  for (int m = 16; m >= 1; m >>= 1) {
    s  += __shfl_xor(s,  m, 32);
    s2 += __shfl_xor(s2, m, 32);
  }
  float mean = s * (1.f / CIN);
  float var  = s2 * (1.f / CIN) - mean * mean;
  float r    = rsqrtf(var + 1e-5f);

  int tm = p >> 4, prow = p & 15;
  int half = (lane >> 3) & 1;                  // sub == lane here
  int e    = (lane & 7) | ((lane >> 4) << 3);
#pragma unroll
  for (int i = 0; i < 8; ++i) {                // kt == i
    int c = lane + i * 32;
    xnp[(((size_t)tm * KT_QKV + i) * 32 + half * 16 + prow) * 16 + e] =
        (__bf16)((vals[i] - mean) * r * g[c]);
  }
}

// ---------------------------------------------------------------------------
// 2) pack weights into B-fragment layout (one-time, tiny)
//    B[((kt*NT + tn)*32 + lane)*16 + e] = W[o = tn*16+(lane&15)]
//                                          [k = kt*32+(lane>>4)*8+(e&7)+(e>>3)*16]
// ---------------------------------------------------------------------------
__global__ void __launch_bounds__(256) cvt_weights(const float* __restrict__ wqkv,
                                                   const float* __restrict__ wout,
                                                   __bf16* __restrict__ wqkvP,
                                                   __bf16* __restrict__ woP) {
  int i = blockIdx.x * blockDim.x + threadIdx.x;
  if (i < KT_QKV * NT_QKV * 512) {
    int e = i & 15, lane = (i >> 4) & 31, t = i >> 9;
    int tn = t % NT_QKV, kt = t / NT_QKV;
    int o = tn * 16 + (lane & 15);
    int k = kt * 32 + (lane >> 4) * 8 + (e & 7) + (e >> 3) * 16;
    wqkvP[i] = (__bf16)wqkv[(size_t)o * CIN + k];
  }
  if (i < KT_OUT * NT_OUT * 512) {
    int e = i & 15, lane = (i >> 4) & 31, t = i >> 9;
    int tn = t % NT_OUT, kt = t / NT_OUT;
    int o = tn * 16 + (lane & 15);
    int k = kt * 32 + (lane >> 4) * 8 + (e & 7) + (e >> 3) * 16;
    woP[i] = (__bf16)wout[(size_t)o * HID + k];
  }
}

// ---------------------------------------------------------------------------
// 3) QKV GEMM: packed fragments; each wave computes 32x64 (2 m x 4 n tiles)
//    -> 8 WMMAs per k-step from 6 b128-pair loads
// ---------------------------------------------------------------------------
__global__ void __launch_bounds__(256) qkv_gemm(const __bf16* __restrict__ Ap,
                                                const __bf16* __restrict__ Bp,
                                                float* __restrict__ Co) {
  int lane = threadIdx.x & 31;
  int wv   = threadIdx.x >> 5;
  int tn0  = (blockIdx.x * 8 + wv) * 4;   // grid.x = 3 -> 96 n-tiles
  int tm0  = blockIdx.y * 2;              // grid.y = 288 -> 576 m-tiles
  v8f acc[2][4] = {};
  for (int kt = 0; kt < KT_QKV; ++kt) {
    v16bf a0 = *(const v16bf*)(Ap + (((size_t)(tm0 + 0) * KT_QKV + kt) * 32 + lane) * 16);
    v16bf a1 = *(const v16bf*)(Ap + (((size_t)(tm0 + 1) * KT_QKV + kt) * 32 + lane) * 16);
#pragma unroll
    for (int j = 0; j < 4; ++j) {
      v16bf b = *(const v16bf*)(Bp + (((size_t)kt * NT_QKV + tn0 + j) * 32 + lane) * 16);
      acc[0][j] = wmma_bf16(a0, b, acc[0][j]);
      acc[1][j] = wmma_bf16(a1, b, acc[1][j]);
    }
  }
  int row = lane & 15, half = lane >> 4;
#pragma unroll
  for (int mi = 0; mi < 2; ++mi)
#pragma unroll
    for (int j = 0; j < 4; ++j)
#pragma unroll
      for (int r = 0; r < 8; ++r) {
        int m = (tm0 + mi) * 16 + r + half * 8;
        int n = (tn0 + j) * 16 + row;
        Co[(size_t)m * QKVO + n] = acc[mi][j][r];
      }
}

// ---------------------------------------------------------------------------
// 4) depthwise 3x3 conv + RoPE + q-scale + bf16 pack to [b,h,s,d]
// ---------------------------------------------------------------------------
__device__ __forceinline__ float dconv(const float* __restrict__ qkv, int b,
                                       int y, int x, int ch, int part,
                                       const float* __restrict__ w) {
  float s = 0.f;
  const float* wc = w + ch * 9;
#pragma unroll
  for (int ky = -1; ky <= 1; ++ky) {
    int yy = y + ky; if ((unsigned)yy >= (unsigned)HH_) continue;
#pragma unroll
    for (int kx = -1; kx <= 1; ++kx) {
      int xx = x + kx; if ((unsigned)xx >= (unsigned)WW_) continue;
      int p = b * SLEN + yy * WW_ + xx;
      s += qkv[(size_t)p * QKVO + part * HID + ch] * wc[(ky + 1) * 3 + (kx + 1)];
    }
  }
  return s;
}

__global__ void __launch_bounds__(256) dwconv_rope(const float* __restrict__ qkv,
                                                   const float* __restrict__ wq,
                                                   const float* __restrict__ wk,
                                                   const float* __restrict__ wvv,
                                                   __bf16* __restrict__ Q,
                                                   __bf16* __restrict__ K,
                                                   __bf16* __restrict__ V) {
  int t  = blockIdx.x * blockDim.x + threadIdx.x;
  int d  = t & 31;
  int r1 = t >> 5;
  int hh = r1 & 7;
  int r2 = r1 >> 3;
  int sp = r2 % SLEN;
  int b  = r2 / SLEN;
  int y = sp / WW_, x = sp % WW_;
  int c1 = hh * DH + d, c2 = c1 + 32;

  float q1 = dconv(qkv, b, y, x, c1, 0, wq);
  float q2 = dconv(qkv, b, y, x, c2, 0, wq);
  float k1 = dconv(qkv, b, y, x, c1, 1, wk);
  float k2 = dconv(qkv, b, y, x, c2, 1, wk);
  float v1 = dconv(qkv, b, y, x, c1, 2, wvv);
  float v2 = dconv(qkv, b, y, x, c2, 2, wvv);

  float invf = __powf(10000.f, -(float)d * (1.f / 32.f));
  float ang  = (float)sp * invf;
  float cs = __cosf(ang), sn = __sinf(ang);
  float qr1 = q1 * cs - q2 * sn, qr2 = q2 * cs + q1 * sn;
  float kr1 = k1 * cs - k2 * sn, kr2 = k2 * cs + k1 * sn;
  const float SC = 0.125f;

  size_t base = ((size_t)(b * HEADS + hh) * SLEN + sp) * DH;
  Q[base + d]      = (__bf16)(qr1 * SC);
  Q[base + d + 32] = (__bf16)(qr2 * SC);
  K[base + d]      = (__bf16)kr1;
  K[base + d + 32] = (__bf16)kr2;
  V[base + d]      = (__bf16)v1;
  V[base + d + 32] = (__bf16)v2;
}

// ---------------------------------------------------------------------------
// 5) flash attention: 4 waves = 64 q rows of one (b,h)
//    K double-buffered via async global->LDS; V transposed into LDS so the
//    P@V B-fragments are single 32B ds loads. Output written in packed
//    A-fragment layout for the final projection GEMM.
// ---------------------------------------------------------------------------
__global__ void __launch_bounds__(128) attn_kernel(const __bf16* __restrict__ Q,
                                                   const __bf16* __restrict__ K,
                                                   const __bf16* __restrict__ V,
                                                   __bf16* __restrict__ AO) {
  __shared__ __bf16 Klds[2][32 * DH];     // [buf][j*64 + d]
  __shared__ __bf16 Vt[2][DH * 32];       // [buf][d*32 + j]  (transposed)
  __shared__ __bf16 Plds[4][16 * 32];

  int lane = threadIdx.x & 31;
  int wv   = threadIdx.x >> 5;
  int bh   = blockIdx.y;                  // 0..31
  int b = bh >> 3, hh = bh & 7;
  int q0 = blockIdx.x * 64 + wv * 16;
  const __bf16* Qb = Q + (size_t)bh * SLEN * DH;
  const __bf16* Kb = K + (size_t)bh * SLEN * DH;
  const __bf16* Vb = V + (size_t)bh * SLEN * DH;

  int row = lane & 15, half = lane >> 4;

  auto load_block = [&](int kv0, int buf) {
    // K rows 32x64 bf16 -> async b128 per lane (256 x 16B chunks)
#pragma unroll
    for (int i = 0; i < 2; ++i) {
      int idx = threadIdx.x + i * 128;
      const __bf16* gsrc = Kb + (size_t)kv0 * DH + idx * 8;
      __bf16* ldst = &Klds[buf][idx * 8];
#if defined(HAVE_ASYNC_LDS)
      __builtin_amdgcn_global_load_async_to_lds_b128(
          as_global_v4(gsrc), as_lds_v4(ldst), 0, 0);
#else
      *(v8bf*)ldst = *(const v8bf*)gsrc;
#endif
    }
    // V rows -> transposed store Vt[d][j]
#pragma unroll
    for (int i = 0; i < 2; ++i) {
      int idx = threadIdx.x + i * 128;
      int j  = idx >> 3;                  // kv row 0..31
      int d0 = (idx & 7) * 8;
      v8bf vv = *(const v8bf*)(Vb + ((size_t)kv0 + j) * DH + d0);
#pragma unroll
      for (int e = 0; e < 8; ++e) Vt[buf][(d0 + e) * 32 + j] = vv[e];
    }
  };

  // resident Q A-fragments (K-dim 0..31, 32..63)
  v16bf qa[2];
#pragma unroll
  for (int kk = 0; kk < 2; ++kk) {
    const __bf16* qp = Qb + (size_t)(q0 + row) * DH + kk * 32 + half * 16;
    v8bf lo = *(const v8bf*)qp;
    v8bf hi = *(const v8bf*)(qp + 8);
#pragma unroll
    for (int e = 0; e < 8; ++e) { qa[kk][e] = lo[e]; qa[kk][8 + e] = hi[e]; }
  }

  v8f Oacc[4] = {};
  float mrow[8], lrow[8];
#pragma unroll
  for (int r = 0; r < 8; ++r) { mrow[r] = -1e30f; lrow[r] = 0.f; }

  load_block(0, 0);
  wait_async0();
  __syncthreads();

  for (int blk = 0; blk < SLEN / 32; ++blk) {
    int cur = blk & 1;
    if (blk + 1 < SLEN / 32) load_block((blk + 1) * 32, cur ^ 1);

    // sim: Q(16x64) @ K^T(64x32) -> two 16x16 C tiles
    v8f stile[2];
#pragma unroll
    for (int nt = 0; nt < 2; ++nt) {
      v8f acc = {};
#pragma unroll
      for (int kk = 0; kk < 2; ++kk) {
        v16bf bfrag = *(const v16bf*)(&Klds[cur][(nt * 16 + row) * DH + kk * 32 + half * 16]);
        acc = wmma_bf16(qa[kk], bfrag, acc);
      }
      stile[nt] = acc;
    }

    // online softmax; row r lives across the 16-lane half-wave
    float scale_r[8];
#pragma unroll
    for (int r = 0; r < 8; ++r) {
      float mx = fmaxf(stile[0][r], stile[1][r]);
#pragma unroll
      for (int m = 8; m >= 1; m >>= 1) mx = fmaxf(mx, __shfl_xor(mx, m, 32));
      float nm = fmaxf(mrow[r], mx);
      float sc = __expf(mrow[r] - nm);
      mrow[r] = nm;
      float p0 = __expf(stile[0][r] - nm);
      float p1 = __expf(stile[1][r] - nm);
      stile[0][r] = p0; stile[1][r] = p1;
      float rs = p0 + p1;
#pragma unroll
      for (int m = 8; m >= 1; m >>= 1) rs += __shfl_xor(rs, m, 32);
      lrow[r] = lrow[r] * sc + rs;
      scale_r[r] = sc;
    }
#pragma unroll
    for (int nt = 0; nt < 4; ++nt)
#pragma unroll
      for (int r = 0; r < 8; ++r) Oacc[nt][r] *= scale_r[r];

    // stage P (C-layout) -> row-major 16x32 for A-fragment reload
#pragma unroll
    for (int r = 0; r < 8; ++r) {
      int prow = r + half * 8;
      Plds[wv][prow * 32 + row]      = (__bf16)stile[0][r];
      Plds[wv][prow * 32 + 16 + row] = (__bf16)stile[1][r];
    }
    asm volatile("s_wait_dscnt 0" ::: "memory");

    v16bf pa;
    {
      const __bf16* pp = &Plds[wv][row * 32 + half * 16];
#pragma unroll
      for (int e = 0; e < 8; ++e) { pa[e] = pp[e]; pa[8 + e] = pp[8 + e]; }
    }

    // O += P(16x32) @ V(32x64): B-fragments are contiguous in Vt
#pragma unroll
    for (int ntd = 0; ntd < 4; ++ntd) {
      v16bf bfrag = *(const v16bf*)(&Vt[cur][(ntd * 16 + row) * 32 + half * 16]);
      Oacc[ntd] = wmma_bf16(pa, bfrag, Oacc[ntd]);
    }

    wait_async0();         // next KV buffer landed
    __syncthreads();       // everyone done with cur before overwrite
  }

  // normalize + store in packed A-fragment layout for out_gemm:
  // AO[((tm*KT_OUT + kt)*32 + lane')*16 + e']
  int tmP = b * (SLEN / 16) + (q0 >> 4);
#pragma unroll
  for (int r = 0; r < 8; ++r) {
    float inv = 1.0f / lrow[r];
    int rowp = r + half * 8;
#pragma unroll
    for (int ntd = 0; ntd < 4; ++ntd) {
      int subc  = (ntd & 1) * 16 + row;          // channel % 32
      int halfp = (subc >> 3) & 1;
      int ep    = (subc & 7) | ((subc >> 4) << 3);
      int kt    = hh * 2 + (ntd >> 1);
      AO[(((size_t)tmP * KT_OUT + kt) * 32 + halfp * 16 + rowp) * 16 + ep] =
          (__bf16)(Oacc[ntd][r] * inv);
    }
  }
}

// ---------------------------------------------------------------------------
// 6) output projection: packed fragments; 2 m x 2 n tiles per wave; + bias
// ---------------------------------------------------------------------------
__global__ void __launch_bounds__(256) out_gemm(const __bf16* __restrict__ Ap,
                                                const __bf16* __restrict__ Bp,
                                                const float* __restrict__ bias,
                                                float* __restrict__ out) {
  int lane = threadIdx.x & 31;
  int wv   = threadIdx.x >> 5;
  int tn0  = wv * 2;                      // 8 waves x 2 = 16 n-tiles
  int tm0  = blockIdx.y * 2;              // grid.y = 288 -> 576 m-tiles
  v8f acc[2][2] = {};
  for (int kt = 0; kt < KT_OUT; ++kt) {
    v16bf a0 = *(const v16bf*)(Ap + (((size_t)(tm0 + 0) * KT_OUT + kt) * 32 + lane) * 16);
    v16bf a1 = *(const v16bf*)(Ap + (((size_t)(tm0 + 1) * KT_OUT + kt) * 32 + lane) * 16);
#pragma unroll
    for (int j = 0; j < 2; ++j) {
      v16bf b = *(const v16bf*)(Bp + (((size_t)kt * NT_OUT + tn0 + j) * 32 + lane) * 16);
      acc[0][j] = wmma_bf16(a0, b, acc[0][j]);
      acc[1][j] = wmma_bf16(a1, b, acc[1][j]);
    }
  }
  int row = lane & 15, half = lane >> 4;
#pragma unroll
  for (int mi = 0; mi < 2; ++mi)
#pragma unroll
    for (int j = 0; j < 2; ++j) {
      int n = (tn0 + j) * 16 + row;
      float bo = bias[n];
#pragma unroll
      for (int r = 0; r < 8; ++r) {
        int m = (tm0 + mi) * 16 + r + half * 8;
        int bb = m / SLEN, sp = m % SLEN;
        out[((size_t)(bb * CIN + n)) * SLEN + sp] = acc[mi][j][r] + bo;
      }
    }
}

// ---------------------------------------------------------------------------
extern "C" void kernel_launch(void* const* d_in, const int* in_sizes, int n_in,
                              void* d_out, int out_size, void* d_ws, size_t ws_size,
                              hipStream_t stream) {
  const float* x      = (const float*)d_in[0];
  const float* norm_g = (const float*)d_in[1];
  const float* w_qkv  = (const float*)d_in[2];
  const float* dw_q   = (const float*)d_in[3];
  const float* dw_k   = (const float*)d_in[4];
  const float* dw_v   = (const float*)d_in[5];
  const float* w_out  = (const float*)d_in[6];
  const float* b_out  = (const float*)d_in[7];
  float* out = (float*)d_out;

  char* ws = (char*)d_ws;
  size_t o_xn  = 0;                                     // bf16 packed-A [PTOT][CIN]
  size_t o_wqP = o_xn  + (size_t)PTOT * CIN * 2;        // bf16 packed-B qkv
  size_t o_woP = o_wqP + (size_t)CIN * QKVO * 2;        // bf16 packed-B out
  size_t o_qkv = o_woP + (size_t)HID * CIN * 2;         // f32  [PTOT][QKVO]
  size_t o_Q   = o_qkv + (size_t)PTOT * QKVO * 4;       // bf16 [B*H][SLEN][DH]
  size_t o_K   = o_Q   + (size_t)PTOT * HID * 2;
  size_t o_V   = o_K   + (size_t)PTOT * HID * 2;
  size_t o_AO  = o_V   + (size_t)PTOT * HID * 2;        // bf16 packed-A [PTOT][HID]

  __bf16* xn    = (__bf16*)(ws + o_xn);
  __bf16* wqkvP = (__bf16*)(ws + o_wqP);
  __bf16* woP   = (__bf16*)(ws + o_woP);
  float*  qkv   = (float*)(ws + o_qkv);
  __bf16* Qb    = (__bf16*)(ws + o_Q);
  __bf16* Kb    = (__bf16*)(ws + o_K);
  __bf16* Vb    = (__bf16*)(ws + o_V);
  __bf16* AO    = (__bf16*)(ws + o_AO);

  ln_kernel<<<PTOT / 8, 256, 0, stream>>>(x, norm_g, xn);
  cvt_weights<<<(KT_QKV * NT_QKV * 512 + 255) / 256, 256, 0, stream>>>(
      w_qkv, w_out, wqkvP, woP);
  qkv_gemm<<<dim3(3, PTOT / 32), 256, 0, stream>>>(xn, wqkvP, qkv);
  dwconv_rope<<<(BB * SLEN * HEADS * 32) / 256, 256, 0, stream>>>(
      qkv, dw_q, dw_k, dw_v, Qb, Kb, Vb);
  attn_kernel<<<dim3(SLEN / 64, BB * HEADS), 128, 0, stream>>>(Qb, Kb, Vb, AO);
  out_gemm<<<dim3(1, PTOT / 32), 256, 0, stream>>>(AO, woP, b_out, out);
}
